// LRU_Attention_LLM_Embedding_35931696398828
// MI455X (gfx1250) — matrix operand order repaired and checked by reading, not compiled, not run on hardware
//
#include <hip/hip_runtime.h>
#include <hip/hip_bf16.h>
#include <stdint.h>

// ---------------------------------------------------------------------------
// Problem constants (from reference setup_inputs)
// ---------------------------------------------------------------------------
#define B_DIM 64
#define S_DIM 512
#define N_TOK (B_DIM * S_DIM)   // 32768 tokens
#define D_DIM 768
#define T_DIM 1536
#define EPS_NORM 1e-12f
#define EPS_LN   1e-5f

typedef __bf16 bf16_t;
typedef __attribute__((ext_vector_type(16))) __bf16 v16bf;
typedef __attribute__((ext_vector_type(8)))  float  v8f;

// ---------------------------------------------------------------------------
// Async global->LDS copy (CDNA5 GLOBAL_LOAD_ASYNC_TO_LDS_B128, ASYNCcnt).
// Builtin signature (from clang diagnostic): (v4i AS1*, v4i AS3*, Imm, Imm).
// Falls back to a synchronous copy if the builtin is unavailable.
// ---------------------------------------------------------------------------
#if defined(__has_builtin)
#if __has_builtin(__builtin_amdgcn_global_load_async_to_lds_b128) && \
    __has_builtin(__builtin_amdgcn_s_wait_asynccnt)
#define USE_ASYNC_LDS 1
#endif
#endif
#ifndef USE_ASYNC_LDS
#define USE_ASYNC_LDS 0
#endif

typedef int v4i_t __attribute__((__vector_size__(16)));
typedef __attribute__((address_space(1))) v4i_t gbl_v4i;
typedef __attribute__((address_space(3))) v4i_t lds_v4i;

__device__ __forceinline__ void cp_async_b128(void* lds_dst, const void* gsrc) {
#if USE_ASYNC_LDS
  __builtin_amdgcn_global_load_async_to_lds_b128(
      (gbl_v4i*)gsrc, (lds_v4i*)lds_dst, 0, 0);
#else
  *(uint4*)lds_dst = *(const uint4*)gsrc;
#endif
}

__device__ __forceinline__ void cp_async_wait_all() {
#if USE_ASYNC_LDS
  __builtin_amdgcn_s_wait_asynccnt(0);
#endif
}

// ---------------------------------------------------------------------------
// WMMA fragment loaders (wave32), contiguous k order k = 16*(L/16) + s for both A and B.
// ---------------------------------------------------------------------------
// 16-bit A matrix 16x32 (MxK): lane L holds row M = L%16; VGPR v holds K pair:
//   K = 2v + 16*(L/16)  (contiguous order, matches the B loaders)
__device__ __forceinline__ v16bf frag_a_lds(const bf16_t* tile, int rs, int lane) {
  const int row = lane & 15, half = lane >> 4;
  const bf16_t* rp = tile + row * rs;
  v16bf a;
#pragma unroll
  for (int v = 0; v < 8; ++v) {
    const int kb = 2 * v + half * 16;
    a[2 * v]     = rp[kb];
    a[2 * v + 1] = rp[kb + 1];
  }
  return a;
}

// 16-bit B matrix 32x16 (KxN), tile stored column-major of B: tile[n*rs + k].
__device__ __forceinline__ v16bf frag_b_colmajor(const bf16_t* tile, int rs, int lane) {
  const int n = lane & 15, half = lane >> 4;
  const bf16_t* np = tile + n * rs;
  v16bf b;
#pragma unroll
  for (int v = 0; v < 8; ++v) {
    const int k = 2 * v + half * 16;
    b[2 * v]     = np[k];
    b[2 * v + 1] = np[k + 1];
  }
  return b;
}

// Same B fragment but tile stored row-major of B: tile[k*rs + n].
__device__ __forceinline__ v16bf frag_b_rowmajor(const bf16_t* tile, int rs, int lane) {
  const int n = lane & 15, half = lane >> 4;
  v16bf b;
#pragma unroll
  for (int v = 0; v < 8; ++v) {
    const int k = 2 * v + half * 16;
    b[2 * v]     = tile[k * rs + n];
    b[2 * v + 1] = tile[(k + 1) * rs + n];
  }
  return b;
}

// ---------------------------------------------------------------------------
// Kernel 0a: zero the per-row sum-of-squares accumulator
// ---------------------------------------------------------------------------
__global__ void init_norms_kernel(float* __restrict__ norms) {
  const int i = blockIdx.x * blockDim.x + threadIdx.x;
  if (i < N_TOK) norms[i] = 0.0f;
}

// ---------------------------------------------------------------------------
// Kernel 0b: convert fc_w (fp32) to bf16 once, enabling async B-tile streaming
// ---------------------------------------------------------------------------
__global__ __launch_bounds__(256) void cvt_w_kernel(
    const float* __restrict__ w, bf16_t* __restrict__ wbf) {
  const int i = (blockIdx.x * 256 + threadIdx.x) * 4;   // D*T / 4 threads total
  const float4 f = *(const float4*)(w + i);
  bf16_t* d = wbf + i;
  d[0] = (bf16_t)f.x; d[1] = (bf16_t)f.y; d[2] = (bf16_t)f.z; d[3] = (bf16_t)f.w;
}

// ---------------------------------------------------------------------------
// Kernel 1: text = gather(text_table, x) @ fc_w^T + fc_b   (bf16 out)
//           + accumulate per-row sum of squares for the L2 norm.
// BM=64 x BN=64 tiles, K step 32.  B tiles (bf16 weights) are async-streamed
// into double-buffered LDS; A tiles are gathered fp32 -> bf16 (overlaps).
// ---------------------------------------------------------------------------
__global__ __launch_bounds__(256) void fc_gemm_kernel(
    const int*    __restrict__ x,
    const float*  __restrict__ text_table,
    const bf16_t* __restrict__ wbf,
    const float*  __restrict__ fc_b,
    bf16_t*       __restrict__ text_bf,
    float*        __restrict__ norms) {
  constexpr int BM = 64, BN = 64, BK = 32;
  constexpr int RS   = BK + 8;        // padded LDS row stride
  constexpr int RSC  = BN + 4;
  constexpr int BBUF = BN * RS;       // one B buffer (2560 bf16 = 5120B)
  __shared__ bf16_t sA[BM * RS];
  __shared__ bf16_t sB[2 * BBUF];
  __shared__ int    sTok[BM];
  __shared__ float  sC[BM * RSC];
  __shared__ float  sRed[BM * 4];

  const int tid  = threadIdx.x;
  const int lane = tid & 31;
  const int wave = tid >> 5;
  const int wm   = wave >> 1;   // 0..3 (M)
  const int wn   = wave & 1;    // 0..1 (N)
  const int m0   = blockIdx.y * BM;
  const int n0   = blockIdx.x * BN;

  const int lr = tid >> 2;        // 0..63
  const int lc = (tid & 3) * 8;   // 0,8,16,24

  // Issue first async B tile (k0 = 0) before anything else.
  cp_async_b128(&sB[lr * RS + lc], wbf + (size_t)(n0 + lr) * T_DIM + lc);

  if (tid < BM) sTok[tid] = x[m0 + tid];
  __syncthreads();

  v8f acc0 = {};
  v8f acc1 = {};

  for (int k0 = 0; k0 < T_DIM; k0 += BK) {
    const int kt  = k0 / BK;
    {   // A tile: gathered text_table rows, fp32 -> bf16 (overlaps async B)
      const float* src = text_table + (size_t)sTok[lr] * T_DIM + (k0 + lc);
      const float4 f0 = ((const float4*)src)[0];
      const float4 f1 = ((const float4*)src)[1];
      bf16_t* dst = &sA[lr * RS + lc];
      dst[0] = (bf16_t)f0.x; dst[1] = (bf16_t)f0.y; dst[2] = (bf16_t)f0.z; dst[3] = (bf16_t)f0.w;
      dst[4] = (bf16_t)f1.x; dst[5] = (bf16_t)f1.y; dst[6] = (bf16_t)f1.z; dst[7] = (bf16_t)f1.w;
      if (k0 + BK < T_DIM) __builtin_prefetch(src + BK, 0, 1);  // global_prefetch_b8
    }
    cp_async_wait_all();
    __syncthreads();
    if (k0 + BK < T_DIM) {  // stream next B tile while this one computes
      cp_async_b128(&sB[((kt + 1) & 1) * BBUF + lr * RS + lc],
                    wbf + (size_t)(n0 + lr) * T_DIM + (k0 + BK + lc));
    }
    const bf16_t* bufB = &sB[(kt & 1) * BBUF];
    const v16bf a  = frag_a_lds(&sA[(wm * 16) * RS], RS, lane);
    const v16bf b0 = frag_b_colmajor(&bufB[(wn * 32) * RS], RS, lane);
    const v16bf b1 = frag_b_colmajor(&bufB[(wn * 32 + 16) * RS], RS, lane);
    acc0 = __builtin_amdgcn_wmma_f32_16x16x32_bf16(false, a, false, b0, (short)0, acc0, false, false);
    acc1 = __builtin_amdgcn_wmma_f32_16x16x32_bf16(false, a, false, b1, (short)0, acc1, false, false);
    __syncthreads();
  }

  // Epilogue: bias, bf16 store, stage fp32 tile for the sum-of-squares.
  const int half = lane >> 4;
  const int cn   = lane & 15;
  const int c0_  = wn * 32 + cn;
  const int c1_  = wn * 32 + 16 + cn;
  const float bias0 = fc_b[n0 + c0_];
  const float bias1 = fc_b[n0 + c1_];
#pragma unroll
  for (int v = 0; v < 8; ++v) {
    const int r = wm * 16 + v + half * 8;          // C layout: M = v + 8*(lane/16)
    const float v0 = acc0[v] + bias0;
    const float v1 = acc1[v] + bias1;
    sC[r * RSC + c0_] = v0;
    sC[r * RSC + c1_] = v1;
    text_bf[(size_t)(m0 + r) * D_DIM + (n0 + c0_)] = (bf16_t)v0;
    text_bf[(size_t)(m0 + r) * D_DIM + (n0 + c1_)] = (bf16_t)v1;
  }
  __syncthreads();
  {
    const int r = tid >> 2, sub = tid & 3;
    float s = 0.0f;
#pragma unroll
    for (int c = 0; c < 16; ++c) {
      const float t = sC[r * RSC + (sub * 16 + c)];
      s += t * t;
    }
    sRed[r * 4 + sub] = s;
  }
  __syncthreads();
  if ((tid & 3) == 0) {
    const int r = tid >> 2;
    atomicAdd(&norms[m0 + r],
              sRed[r * 4] + sRed[r * 4 + 1] + sRed[r * 4 + 2] + sRed[r * 4 + 3]);
  }
}

// ---------------------------------------------------------------------------
// Kernel 2: L2-normalize each text row in place (bf16)
// ---------------------------------------------------------------------------
__global__ __launch_bounds__(256) void scale_rows_kernel(
    bf16_t* __restrict__ text_bf, const float* __restrict__ norms) {
  const int row = blockIdx.x;
  const float s = 1.0f / fmaxf(sqrtf(norms[row]), EPS_NORM);
  bf16_t* rp = text_bf + (size_t)row * D_DIM;
  for (int c = threadIdx.x; c < D_DIM; c += 256)
    rp[c] = (bf16_t)((float)rp[c] * s);
}

// ---------------------------------------------------------------------------
// Kernel 3: attention. One block = 32 query rows x all 512 keys of one batch.
// Scores live in LDS (exact two-pass softmax; 320KB WGP LDS). K and V tiles
// are async-DMA'd (GLOBAL_LOAD_ASYNC_TO_LDS_B128) into ping-pong buffers so
// the next tile streams while the current one is in the WMMAs.
// ---------------------------------------------------------------------------
__global__ __launch_bounds__(256) void attn_kernel(
    const int*    __restrict__ x,
    const float*  __restrict__ token_emb,
    const bf16_t* __restrict__ text_bf,
    const float*  __restrict__ alpha,
    float*        __restrict__ blend) {
  constexpr int QT   = 32;
  constexpr int RS_Q = D_DIM + 8;   // 776
  constexpr int RS_K = 40;
  constexpr int RS_S = S_DIM + 4;   // 516
  constexpr int RS_P = S_DIM + 8;   // 520
  constexpr int RS_V = 72;
  constexpr int KBUF = 64 * RS_K;   // one K/V buffer: 2560 bf16 = 5120 B

  constexpr int OFF_TOK = 0;
  constexpr int OFF_Q   = 128;
  constexpr int OFF_K   = OFF_Q + QT * RS_Q * 2;     // 49792
  constexpr int OFF_S   = OFF_K + 2 * KBUF * 2;      // 60032 (double buffer)
  constexpr int OFF_RED = OFF_S + QT * RS_S * 4;     // 126080
  constexpr int SMEM_SZ = OFF_RED + QT * 8 * 4;      // 127104 B (< 320KB WGP LDS)
  __shared__ alignas(16) unsigned char smem[SMEM_SZ];
  int*    sTok = (int*)(smem + OFF_TOK);
  bf16_t* sQ   = (bf16_t*)(smem + OFF_Q);
  bf16_t* sKV  = (bf16_t*)(smem + OFF_K);
  float*  sS   = (float*)(smem + OFF_S);
  bf16_t* sP   = (bf16_t*)(smem + OFF_Q);     // aliases Q (dead after phase 1)
  float*  sRed = (float*)(smem + OFF_RED);

  const int tid  = threadIdx.x;
  const int lane = tid & 31;
  const int wave = tid >> 5;
  const int wm   = wave >> 2;   // 0..1
  const int wn   = wave & 3;    // 0..3
  const int b    = blockIdx.y;
  const int q0   = blockIdx.x * QT;
  const int rowbase = b * S_DIM;

  const int kr = tid >> 2;         // K-tile thread map: 64 rows x 32 cols
  const int kc = (tid & 3) * 8;
  const int vr = tid >> 3;         // V-tile thread map: 32 rows x 64 cols
  const int vc = (tid & 7) * 8;

  // Issue first K tile (nk=0, k0=0) immediately; overlaps the Q gather below.
  cp_async_b128(&sKV[kr * RS_K + kc],
                text_bf + (size_t)(rowbase + kr) * D_DIM + kc);

  if (tid < QT) sTok[tid] = x[rowbase + q0 + tid];
  __syncthreads();

  {   // Preload Q tile (item = gathered token_emb rows) as bf16: 32 x 768
    const int r = tid >> 3;            // 0..31
    const int cbase = (tid & 7) * 96;  // 8 threads per row
    const float* src = token_emb + (size_t)sTok[r] * D_DIM + cbase;
    bf16_t* dst = &sQ[r * RS_Q + cbase];
#pragma unroll
    for (int i = 0; i < 24; ++i) {
      const float4 f = ((const float4*)src)[i];
      dst[4 * i]     = (bf16_t)f.x; dst[4 * i + 1] = (bf16_t)f.y;
      dst[4 * i + 2] = (bf16_t)f.z; dst[4 * i + 3] = (bf16_t)f.w;
    }
  }

  const float inv_sqrt_d = rsqrtf((float)D_DIM);

  // ---- Phase 1: S = Q K^T / sqrt(D); 192 K tiles (8 nk x 24 kk) ----------
  for (int nk = 0; nk < S_DIM / 64; ++nk) {
    v8f acc = {};
    for (int kk = 0; kk < 24; ++kk) {
      const int idx = nk * 24 + kk;
      cp_async_wait_all();
      __syncthreads();                       // tile idx landed; prior reads done
      if (idx + 1 < 192) {                   // stream tile idx+1
        const int nidx = idx + 1;
        const int nnk = nidx / 24, nkk = nidx % 24;
        cp_async_b128(&sKV[(nidx & 1) * KBUF + kr * RS_K + kc],
                      text_bf + (size_t)(rowbase + nnk * 64 + kr) * D_DIM +
                          (nkk * 32 + kc));
      }
      const bf16_t* bufK = &sKV[(idx & 1) * KBUF];
      const v16bf a  = frag_a_lds(&sQ[(wm * 16) * RS_Q + kk * 32], RS_Q, lane);
      const v16bf bb = frag_b_colmajor(&bufK[(wn * 16) * RS_K], RS_K, lane);
      acc = __builtin_amdgcn_wmma_f32_16x16x32_bf16(false, a, false, bb, (short)0, acc, false, false);
      __syncthreads();
    }
    const int half = lane >> 4, cn = lane & 15;
#pragma unroll
    for (int v = 0; v < 8; ++v) {
      const int r = wm * 16 + v + half * 8;
      const int c = nk * 64 + wn * 16 + cn;
      sS[r * RS_S + c] = acc[v] * inv_sqrt_d;
    }
  }
  __syncthreads();

  // ---- Phase 2: exact softmax over the full 512-key row ------------------
  const int srow = tid >> 3;    // 0..31, 8 threads per row
  const int ssub = tid & 7;
  {
    float lmax = -3.4e38f;
    for (int c = ssub; c < S_DIM; c += 8) lmax = fmaxf(lmax, sS[srow * RS_S + c]);
    sRed[srow * 8 + ssub] = lmax;
  }
  __syncthreads();
  float rmax = sRed[srow * 8];
#pragma unroll
  for (int i = 1; i < 8; ++i) rmax = fmaxf(rmax, sRed[srow * 8 + i]);
  __syncthreads();
  {
    float lsum = 0.0f;
    for (int c = ssub; c < S_DIM; c += 8) {
      const float e = __expf(sS[srow * RS_S + c] - rmax);
      sS[srow * RS_S + c] = e;
      lsum += e;
    }
    sRed[srow * 8 + ssub] = lsum;
  }
  __syncthreads();
  float rsum = 0.0f;
#pragma unroll
  for (int i = 0; i < 8; ++i) rsum += sRed[srow * 8 + i];
  const float rinv = 1.0f / rsum;
  __syncthreads();
  for (int c = ssub; c < S_DIM; c += 8)
    sP[srow * RS_P + c] = (bf16_t)(sS[srow * RS_S + c] * rinv);
  __syncthreads();

  // ---- Phase 3: O = P V, fused blend with alpha; 192 V tiles (12 x 16) ---
  // Issue first V tile (d0=0, t0=0); K buffers are free after phase 1.
  cp_async_b128(&sKV[vr * RS_V + vc],
                text_bf + (size_t)(rowbase + vr) * D_DIM + vc);

  const float a_ = alpha[0];
  const float one_m_a = 1.0f - a_;
  for (int dc = 0; dc < 12; ++dc) {
    v8f acc = {};
    for (int tt = 0; tt < 16; ++tt) {
      const int idx = dc * 16 + tt;
      cp_async_wait_all();
      __syncthreads();
      if (idx + 1 < 192) {
        const int nidx = idx + 1;
        const int nd = nidx / 16, nt = nidx % 16;
        cp_async_b128(&sKV[(nidx & 1) * KBUF + vr * RS_V + vc],
                      text_bf + (size_t)(rowbase + nt * 32 + vr) * D_DIM +
                          (nd * 64 + vc));
      }
      const bf16_t* bufV = &sKV[(idx & 1) * KBUF];
      const v16bf a  = frag_a_lds(&sP[(wm * 16) * RS_P + tt * 32], RS_P, lane);
      const v16bf bb = frag_b_rowmajor(&bufV[wn * 16], RS_V, lane);
      acc = __builtin_amdgcn_wmma_f32_16x16x32_bf16(false, a, false, bb, (short)0, acc, false, false);
      __syncthreads();
    }
    const int half = lane >> 4, cn = lane & 15;
#pragma unroll
    for (int v = 0; v < 8; ++v) {
      const int r  = wm * 16 + v + half * 8;
      const int c  = dc * 64 + wn * 16 + cn;
      const int gn = rowbase + q0 + r;
      const float item = token_emb[(size_t)sTok[r] * D_DIM + c];
      blend[(size_t)gn * D_DIM + c] = one_m_a * item + a_ * acc[v];
    }
  }
}

// ---------------------------------------------------------------------------
// Kernel 4: LayerNorm over D=768 per token + mask output
// ---------------------------------------------------------------------------
__global__ __launch_bounds__(256) void ln_kernel(
    const float* __restrict__ blend, const int* __restrict__ x,
    const float* __restrict__ gamma, const float* __restrict__ beta,
    float* __restrict__ out, float* __restrict__ mask_out) {
  __shared__ float red[256];
  const int n   = blockIdx.x;
  const int tid = threadIdx.x;
  const float* rp = blend + (size_t)n * D_DIM;
  const float v0 = rp[tid], v1 = rp[tid + 256], v2 = rp[tid + 512];
  red[tid] = v0 + v1 + v2;
  __syncthreads();
  for (int off = 128; off > 0; off >>= 1) {
    if (tid < off) red[tid] += red[tid + off];
    __syncthreads();
  }
  const float mu = red[0] * (1.0f / D_DIM);
  __syncthreads();
  const float d0 = v0 - mu, d1 = v1 - mu, d2 = v2 - mu;
  red[tid] = d0 * d0 + d1 * d1 + d2 * d2;
  __syncthreads();
  for (int off = 128; off > 0; off >>= 1) {
    if (tid < off) red[tid] += red[tid + off];
    __syncthreads();
  }
  const float inv = rsqrtf(red[0] * (1.0f / D_DIM) + EPS_LN);
  float* op = out + (size_t)n * D_DIM;
  op[tid]       = d0 * inv * gamma[tid]       + beta[tid];
  op[tid + 256] = d1 * inv * gamma[tid + 256] + beta[tid + 256];
  op[tid + 512] = d2 * inv * gamma[tid + 512] + beta[tid + 512];
  if (tid == 0) mask_out[n] = (x[n] > 0) ? 1.0f : 0.0f;
}

// ---------------------------------------------------------------------------
// Launch pipeline
// ---------------------------------------------------------------------------
extern "C" void kernel_launch(void* const* d_in, const int* in_sizes, int n_in,
                              void* d_out, int out_size, void* d_ws, size_t ws_size,
                              hipStream_t stream) {
  (void)in_sizes; (void)n_in; (void)out_size; (void)ws_size;
  const int*   x          = (const int*)d_in[0];
  const float* token_emb  = (const float*)d_in[1];
  const float* text_table = (const float*)d_in[2];
  const float* fc_w       = (const float*)d_in[3];
  const float* fc_b       = (const float*)d_in[4];
  const float* alpha      = (const float*)d_in[5];
  const float* ln_gamma   = (const float*)d_in[6];
  const float* ln_beta    = (const float*)d_in[7];

  float* out      = (float*)d_out;
  float* mask_out = out + (size_t)N_TOK * D_DIM;

  // Workspace: [blend fp32 N*D][text bf16 N*D][norms fp32 N][wbf bf16 D*T]
  float*  ws_blend = (float*)d_ws;
  bf16_t* ws_text  = (bf16_t*)(ws_blend + (size_t)N_TOK * D_DIM);
  float*  ws_norms = (float*)(ws_text + (size_t)N_TOK * D_DIM);
  bf16_t* ws_wbf   = (bf16_t*)(ws_norms + N_TOK);

  init_norms_kernel<<<N_TOK / 256, 256, 0, stream>>>(ws_norms);
  cvt_w_kernel<<<(D_DIM * T_DIM) / (256 * 4), 256, 0, stream>>>(fc_w, ws_wbf);
  fc_gemm_kernel<<<dim3(D_DIM / 64, N_TOK / 64), 256, 0, stream>>>(
      x, text_table, ws_wbf, fc_b, ws_text, ws_norms);
  scale_rows_kernel<<<N_TOK, 256, 0, stream>>>(ws_text, ws_norms);
  attn_kernel<<<dim3(S_DIM / 32, B_DIM), 256, 0, stream>>>(
      x, token_emb, ws_text, alpha, ws_blend);
  ln_kernel<<<N_TOK, 256, 0, stream>>>(ws_blend, x, ln_gamma, ln_beta, out, mask_out);
}
